// CharLSTM_19112604467264
// MI455X (gfx1250) — compile-verified, matrix operand order checked by reading, not run on hardware
//
#include <hip/hip_runtime.h>
#include <hip/hip_bf16.h>
#include <stdint.h>

typedef __bf16 bf16;
typedef __bf16 v16bf __attribute__((ext_vector_type(16)));
typedef float  v8f   __attribute__((ext_vector_type(8)));

// ---------------- problem constants ----------------
constexpr int Bn  = 64;     // batch
constexpr int Tn  = 511;    // time
constexpr int Un  = 512;    // hidden
constexpr int NG  = 2048;   // 4*U gate width
constexpr int K0  = 640;    // 128 (padded 112-dim x) + 512
constexpr int K12 = 1024;   // 512 + 512
constexpr int TVn = 130;    // vocab out
constexpr int NWG = 32;     // persistent workgroups (u-slices of 16)

// ---------------- workspace layout (bytes) ----------------
constexpr size_t OFF_WT0  = 0;
constexpr size_t SZ_WT0   = (size_t)NG * K0 * 2;
constexpr size_t OFF_WT1  = OFF_WT0 + SZ_WT0;
constexpr size_t SZ_WT12  = (size_t)NG * K12 * 2;
constexpr size_t OFF_WT2  = OFF_WT1 + SZ_WT12;
constexpr size_t OFF_WDT  = OFF_WT2 + SZ_WT12;
constexpr size_t SZ_WDT   = (size_t)144 * Un * 2;          // Wd^T padded 130->144 rows
constexpr size_t OFF_X    = OFF_WDT + SZ_WDT;
constexpr size_t SZ_X     = (size_t)Bn * Tn * 128 * 2;     // padded embeddings, bf16
constexpr size_t OFF_H    = OFF_X + SZ_X;
constexpr size_t SZ_H     = (size_t)3 * 2 * Bn * Un * 2;   // [layer][parity][B][U] bf16
constexpr size_t OFF_OUTS = OFF_H + SZ_H;
constexpr size_t SZ_OUTS  = (size_t)Bn * Tn * Un * 2;      // masked top-layer h, bf16
constexpr size_t OFF_BAR  = OFF_OUTS + SZ_OUTS;            // barrier counter

// ---------------- helpers ----------------
__device__ inline v8f v8f_zero() {
  v8f z = {0.f, 0.f, 0.f, 0.f, 0.f, 0.f, 0.f, 0.f};
  return z;
}

// Load a 16x32 bf16 fragment (A layout; B uses the mirrored identical layout
// when the matrix is stored transposed [N][K]).
// lanes 0-15 : row = lane,     K = k0+[0..7] and k0+[16..23]
// lanes 16-31: row = lane-16,  K = k0+[8..15] and k0+[24..31]
__device__ inline v16bf load_frag(const bf16* __restrict__ base, size_t stride, int k0) {
  const int lane = threadIdx.x & 31;
  const int r    = lane & 15;
  const int hi   = lane >> 4;
  const bf16* p  = base + (size_t)r * stride + (size_t)(k0 + hi * 8);
  union { uint4 u[2]; v16bf v; } tmp;
  tmp.u[0] = *(const uint4*)(p);        // 8 bf16
  tmp.u[1] = *(const uint4*)(p + 16);   // 8 bf16
  return tmp.v;
}

__device__ inline float sigm(float x) { return 1.0f / (1.0f + __expf(-x)); }

// Generation-counter grid barrier across NWG persistent workgroups.
__device__ inline void grid_barrier(unsigned* __restrict__ bar, unsigned target) {
  __threadfence();
  __syncthreads();
  if (threadIdx.x == 0) {
    atomicAdd(bar, 1u);
    while (__hip_atomic_load(bar, __ATOMIC_ACQUIRE, __HIP_MEMORY_SCOPE_AGENT) < target) {
      __builtin_amdgcn_s_sleep(2);
    }
  }
  __syncthreads();
  __threadfence();
}

// ---------------- prep kernels ----------------

// WT[n][k] = stacked+transposed [Wx; Wh] in bf16.  (k < kx -> Wx, kpad..ktot -> Wh)
__global__ void k_wt(const float* __restrict__ Wx, const float* __restrict__ Wh,
                     bf16* __restrict__ WT, int kx, int kpad, int ktot) {
  size_t idx = (size_t)blockIdx.x * blockDim.x + threadIdx.x;
  size_t total = (size_t)NG * ktot;
  if (idx >= total) return;
  int n = (int)(idx / ktot);
  int k = (int)(idx % ktot);
  float v = 0.0f;
  if (k < kx)          v = Wx[(size_t)k * NG + n];
  else if (k >= kpad)  v = Wh[(size_t)(k - kpad) * NG + n];
  WT[idx] = (bf16)v;
}

// WdT[n][k] (144x512), rows >= 130 zero.
__global__ void k_wdt(const float* __restrict__ Wd, bf16* __restrict__ WdT) {
  int idx = blockIdx.x * blockDim.x + threadIdx.x;
  if (idx >= 144 * Un) return;
  int n = idx / Un, k = idx % Un;
  float v = (n < TVn) ? Wd[(size_t)k * TVn + n] : 0.0f;
  WdT[idx] = (bf16)v;
}

// x[b][t][0..127] = concat(rhythm16, meter16, key16, tune64, zero-pad16), bf16
__global__ void k_embed(const int* __restrict__ tune, const int* __restrict__ rhythm,
                        const int* __restrict__ meter, const int* __restrict__ key_sig,
                        const float* __restrict__ E_tune, const float* __restrict__ E_rhythm,
                        const float* __restrict__ E_meter, const float* __restrict__ E_key,
                        bf16* __restrict__ xbuf) {
  const int pos = blockIdx.x;           // b*T + t
  const int c   = threadIdx.x;          // 0..127
  const int b   = pos / Tn;
  float v = 0.0f;
  if (c < 16)       v = E_rhythm[rhythm[b] * 16 + c];
  else if (c < 32)  v = E_meter[meter[b] * 16 + (c - 16)];
  else if (c < 48)  v = E_key[key_sig[b] * 16 + (c - 32)];
  else if (c < 112) v = E_tune[tune[pos] * 64 + (c - 48)];
  xbuf[(size_t)pos * 128 + c] = (bf16)v;
}

__global__ void k_zero(uint32_t* __restrict__ hwords, size_t nwords,
                       unsigned* __restrict__ bar) {
  size_t idx = (size_t)blockIdx.x * blockDim.x + threadIdx.x;
  if (idx < nwords) hwords[idx] = 0u;
  if (idx == 0) *bar = 0u;
}

// ---------------- persistent LSTM kernel ----------------
// 32 WGs x 128 threads. WG owns hidden cols [u0,u0+16) of every layer.
// Wave w owns batch rows [16w,16w+16). Per (t,layer): 64x64 gate tile via
// 4 f32 accumulators (i,f,g,o); K loop is software-pipelined one deep and
// unrolled x2 so the double-buffer rotation is pure register renaming.
__global__ void __launch_bounds__(128, 1)
k_lstm(const bf16* __restrict__ WT0, const bf16* __restrict__ WT1,
       const bf16* __restrict__ WT2,
       const float* __restrict__ bias0, const float* __restrict__ bias1,
       const float* __restrict__ bias2,
       const int* __restrict__ tune_length,
       const bf16* __restrict__ xbuf, bf16* __restrict__ hbuf,
       bf16* __restrict__ outs, unsigned* __restrict__ bar) {
  const int u0   = blockIdx.x * 16;
  const int wave = threadIdx.x >> 5;
  const int lane = threadIdx.x & 31;
  const int r16  = lane & 15;
  const int hi   = lane >> 4;
  const int m0   = wave * 16;
  const int bm   = m0 + hi * 8;   // this lane's 8 batch rows: bm..bm+7

  int len[8];
#pragma unroll
  for (int r = 0; r < 8; ++r) len[r] = tune_length[bm + r];

  const bf16*  WTs[3] = {WT0, WT1, WT2};
  const float* bs[3]  = {bias0, bias1, bias2};
  float bias[3][4];
#pragma unroll
  for (int l = 0; l < 3; ++l)
#pragma unroll
    for (int g = 0; g < 4; ++g)
      bias[l][g] = bs[l][g * Un + u0 + r16];

  v8f c[3], hreg[3];
#pragma unroll
  for (int l = 0; l < 3; ++l) { c[l] = v8f_zero(); hreg[l] = v8f_zero(); }

  unsigned nbar = 0;

#pragma unroll 1
  for (int t = 0; t < Tn; ++t) {
    const int cur = t & 1, prev = cur ^ 1;
#pragma unroll
    for (int l = 0; l < 3; ++l) {
      const int K    = (l == 0) ? K0 : K12;
      const int kbnd = (l == 0) ? 128 : Un;
      const bf16* a0base;
      size_t a0stride;
      if (l == 0) {
        a0base = xbuf + ((size_t)m0 * Tn + t) * 128;   // x_t rows, stride T*128
        a0stride = (size_t)Tn * 128;
      } else {
        a0base = hbuf + (((size_t)(l - 1) * 2 + cur) * Bn + m0) * Un;  // h_{l-1}(t)
        a0stride = Un;
      }
      const bf16* a1base = hbuf + (((size_t)l * 2 + prev) * Bn + m0) * Un; // h_l(t-1)
      const bf16* wt = WTs[l];
      const bf16* wb0 = wt + (size_t)(0 * Un + u0) * K;
      const bf16* wb1 = wt + (size_t)(1 * Un + u0) * K;
      const bf16* wb2 = wt + (size_t)(2 * Un + u0) * K;
      const bf16* wb3 = wt + (size_t)(3 * Un + u0) * K;

      auto load_a = [&](int kc) -> v16bf {
        return (kc < kbnd) ? load_frag(a0base, a0stride, kc)
                           : load_frag(a1base, Un, kc - kbnd);
      };

      v8f acc0 = v8f_zero(), acc1 = v8f_zero(), acc2 = v8f_zero(), acc3 = v8f_zero();

      // software-pipelined K loop, 1 deep; unroll x2 => buffer swap is renaming
      v16bf a_c  = load_a(0);
      v16bf b0_c = load_frag(wb0, K, 0);
      v16bf b1_c = load_frag(wb1, K, 0);
      v16bf b2_c = load_frag(wb2, K, 0);
      v16bf b3_c = load_frag(wb3, K, 0);
#pragma unroll 2
      for (int kk = 0; kk < K; kk += 32) {
        const int kn = (kk + 32 < K) ? (kk + 32) : kk;  // uniform; last reload unused
        v16bf a_n  = load_a(kn);
        v16bf b0_n = load_frag(wb0, K, kn);
        v16bf b1_n = load_frag(wb1, K, kn);
        v16bf b2_n = load_frag(wb2, K, kn);
        v16bf b3_n = load_frag(wb3, K, kn);
        acc0 = __builtin_amdgcn_wmma_f32_16x16x32_bf16(false, a_c, false, b0_c, (short)0, acc0, false, false);
        acc1 = __builtin_amdgcn_wmma_f32_16x16x32_bf16(false, a_c, false, b1_c, (short)0, acc1, false, false);
        acc2 = __builtin_amdgcn_wmma_f32_16x16x32_bf16(false, a_c, false, b2_c, (short)0, acc2, false, false);
        acc3 = __builtin_amdgcn_wmma_f32_16x16x32_bf16(false, a_c, false, b3_c, (short)0, acc3, false, false);
        a_c = a_n; b0_c = b0_n; b1_c = b1_n; b2_c = b2_n; b3_c = b3_n;
      }

      bf16* hout = hbuf + ((size_t)l * 2 + cur) * Bn * Un;
#pragma unroll
      for (int r = 0; r < 8; ++r) {
        const int brow = bm + r;
        float zi = acc0[r] + bias[l][0];
        float zf = acc1[r] + bias[l][1];
        float zg = acc2[r] + bias[l][2];
        float zo = acc3[r] + bias[l][3];
        float ig = sigm(zi), fg = sigm(zf), og = sigm(zo);
        float gg = tanhf(zg);
        float c2 = fg * c[l][r] + ig * gg;
        float h2 = og * tanhf(c2);
        const bool msk = (t < len[r]);
        c2 = msk ? c2 : c[l][r];
        h2 = msk ? h2 : hreg[l][r];
        c[l][r]    = c2;
        hreg[l][r] = h2;
        hout[(size_t)brow * Un + u0 + r16] = (bf16)h2;
        if (l == 2) {
          float ov = msk ? h2 : 0.0f;
          outs[((size_t)brow * Tn + t) * Un + u0 + r16] = (bf16)ov;
        }
      }
      // Only 2 grid barriers per timestep are required: after L0 and L1.
      // L2(t) writes are ordered before L2(t+1) reads by the L0(t+1) barrier.
      if (l < 2) grid_barrier(bar, (unsigned)NWG * (++nbar));
    }
  }
}

// ---------------- projection: out = outs @ Wd + bd ----------------
// M = B*T = 32704 (2044 tiles), N padded 130->144 (9 tiles), K = 512.
__global__ void __launch_bounds__(128)
k_proj(const bf16* __restrict__ outs, const bf16* __restrict__ WdT,
       const float* __restrict__ bd, float* __restrict__ out) {
  const int gw = blockIdx.x * 4 + (threadIdx.x >> 5);
  const int NT = 9;
  const int tm = gw / NT, tn = gw % NT;
  if (tm >= (Bn * Tn) / 16) return;
  const int lane = threadIdx.x & 31;
  const int r16  = lane & 15;
  const int hi   = lane >> 4;
  const int m0 = tm * 16, n0 = tn * 16;

  v8f acc = v8f_zero();
  const bf16* abase = outs + (size_t)m0 * Un;
  const bf16* bbase = WdT + (size_t)n0 * Un;
#pragma unroll 4
  for (int kk = 0; kk < Un; kk += 32) {
    v16bf a  = load_frag(abase, Un, kk);
    v16bf bb = load_frag(bbase, Un, kk);
    acc = __builtin_amdgcn_wmma_f32_16x16x32_bf16(
        false, a, false, bb, (short)0, acc, false, false);
  }
  const int col = n0 + r16;
  if (col < TVn) {
    const float bias = bd[col];
#pragma unroll
    for (int r = 0; r < 8; ++r)
      out[(size_t)(m0 + hi * 8 + r) * TVn + col] = acc[r] + bias;
  }
}

// ---------------- launch ----------------
extern "C" void kernel_launch(void* const* d_in, const int* in_sizes, int n_in,
                              void* d_out, int out_size, void* d_ws, size_t ws_size,
                              hipStream_t stream) {
  (void)in_sizes; (void)n_in; (void)out_size; (void)ws_size;

  const int*   tune     = (const int*)d_in[0];
  const int*   rhythm   = (const int*)d_in[1];
  const int*   meter    = (const int*)d_in[2];
  const int*   key_sig  = (const int*)d_in[3];
  const int*   tune_len = (const int*)d_in[4];
  const float* E_tune   = (const float*)d_in[5];
  const float* E_rhythm = (const float*)d_in[6];
  const float* E_meter  = (const float*)d_in[7];
  const float* E_key    = (const float*)d_in[8];
  const float* Wd       = (const float*)d_in[9];
  const float* bd       = (const float*)d_in[10];
  const float* Wx0 = (const float*)d_in[11];
  const float* Wh0 = (const float*)d_in[12];
  const float* b0  = (const float*)d_in[13];
  const float* Wx1 = (const float*)d_in[14];
  const float* Wh1 = (const float*)d_in[15];
  const float* b1  = (const float*)d_in[16];
  const float* Wx2 = (const float*)d_in[17];
  const float* Wh2 = (const float*)d_in[18];
  const float* b2  = (const float*)d_in[19];
  float* out = (float*)d_out;
  char*  ws  = (char*)d_ws;

  bf16* WT0  = (bf16*)(ws + OFF_WT0);
  bf16* WT1  = (bf16*)(ws + OFF_WT1);
  bf16* WT2  = (bf16*)(ws + OFF_WT2);
  bf16* WdT  = (bf16*)(ws + OFF_WDT);
  bf16* xbuf = (bf16*)(ws + OFF_X);
  bf16* hbuf = (bf16*)(ws + OFF_H);
  bf16* outs = (bf16*)(ws + OFF_OUTS);
  unsigned* bar = (unsigned*)(ws + OFF_BAR);

  k_wt<<<(NG * K0 + 255) / 256, 256, 0, stream>>>(Wx0, Wh0, WT0, 112, 128, K0);
  k_wt<<<(NG * K12 + 255) / 256, 256, 0, stream>>>(Wx1, Wh1, WT1, Un, Un, K12);
  k_wt<<<(NG * K12 + 255) / 256, 256, 0, stream>>>(Wx2, Wh2, WT2, Un, Un, K12);
  k_wdt<<<(144 * Un + 255) / 256, 256, 0, stream>>>(Wd, WdT);
  k_embed<<<Bn * Tn, 128, 0, stream>>>(tune, rhythm, meter, key_sig,
                                       E_tune, E_rhythm, E_meter, E_key, xbuf);
  size_t hwords = SZ_H / 4;
  k_zero<<<(unsigned)((hwords + 255) / 256), 256, 0, stream>>>((uint32_t*)hbuf, hwords, bar);
  k_lstm<<<NWG, 128, 0, stream>>>(WT0, WT1, WT2, b0, b1, b2, tune_len,
                                  xbuf, hbuf, outs, bar);
  int waves = (Bn * Tn / 16) * 9;
  k_proj<<<(waves + 3) / 4, 128, 0, stream>>>(outs, WdT, bd, out);
}
